// QuantizedLinear_6992206758378
// MI455X (gfx1250) — matrix operand order repaired and checked by reading, not compiled, not run on hardware
//
#include <hip/hip_runtime.h>

typedef __attribute__((ext_vector_type(16))) _Float16 v16h;
typedef __attribute__((ext_vector_type(8)))  _Float16 v8h;
typedef __attribute__((ext_vector_type(8)))  float    v8f;
typedef __attribute__((ext_vector_type(4)))  float    v4f;

#define IN_F    4096
#define OUT_F   4096
#define GROUPS  32          // IN_F / 128
#define PACKW   512         // IN_F / 8 packed int32 per output row
#define BM      128
#define BN      128
#define BK      32
#define SK      40          // padded LDS row stride (halves): 80B -> bank-conflict free
#define NKSTEPS (IN_F / BK) // 128

__global__ __launch_bounds__(256)
void w4_gemm_wmma(const float* __restrict__ x,
                  const int*   __restrict__ qw,
                  const float* __restrict__ scales,
                  const float* __restrict__ zeros,
                  const float* __restrict__ bias,
                  float* __restrict__ out)
{
    __shared__ _Float16 As[2][BM * SK];   // activations, f16, [row][k]
    __shared__ _Float16 Bs[2][BN * SK];   // dequantized weights, f16, [outfeat][k]

    const int t    = threadIdx.x;
    const int lane = t & 31;
    const int w    = t >> 5;            // wave 0..7
    const int wm   = (w >> 2) * 64;     // wave M offset inside block tile
    const int wn   = (w & 3) * 32;      // wave N offset inside block tile
    const int bm   = blockIdx.y * BM;   // token offset
    const int bn   = blockIdx.x * BN;   // out-feature offset

    // global-load work split: each thread owns half a row of each tile
    const int arow = t >> 1;
    const int acol = (t & 1) * 16;      // 16 floats of x
    const int brow = t >> 1;
    const int bc   = (t & 1) * 2;       // 2 packed int32 (16 int4)

    const float* aptr0 = x  + (size_t)(bm + arow) * IN_F  + acol;
    const int*   qptr0 = qw + (size_t)(bn + brow) * PACKW + bc;
    const float* sptr  = scales + (size_t)(bn + brow) * GROUPS;
    const float* zptr  = zeros  + (size_t)(bn + brow) * GROUPS;

    v4f   areg[4];
    int2  breg;
    float sreg, zreg;

    auto fetch = [&](int kk) {
        const float* ap = aptr0 + kk * BK;
        areg[0] = *(const v4f*)(ap + 0);
        areg[1] = *(const v4f*)(ap + 4);
        areg[2] = *(const v4f*)(ap + 8);
        areg[3] = *(const v4f*)(ap + 12);
        breg = *(const int2*)(qptr0 + kk * (BK / 8));
        const int g = (kk * BK) >> 7;   // group = k / 128
        sreg = sptr[g];
        zreg = zptr[g];
    };

    auto stage = [&](int b) {
        // x: f32 -> f16
        v8h h0, h1;
#pragma unroll
        for (int i = 0; i < 4; ++i) {
            h0[i]     = (_Float16)areg[0][i];
            h0[4 + i] = (_Float16)areg[1][i];
            h1[i]     = (_Float16)areg[2][i];
            h1[4 + i] = (_Float16)areg[3][i];
        }
        _Float16* ad = &As[b][arow * SK + acol];
        *(v8h*)(ad)     = h0;
        *(v8h*)(ad + 8) = h1;

        // int4 dequant: (q - zero) * scale -> f16
        v8h b0, b1;
#pragma unroll
        for (int i = 0; i < 8; ++i) {
            b0[i] = (_Float16)(((float)((breg.x >> (4 * i)) & 15) - zreg) * sreg);
            b1[i] = (_Float16)(((float)((breg.y >> (4 * i)) & 15) - zreg) * sreg);
        }
        _Float16* bd = &Bs[b][brow * SK + bc * 8];
        *(v8h*)(bd)     = b0;
        *(v8h*)(bd + 8) = b1;
    };

    v8f acc[4][2];
    const v8f zero8 = {};
#pragma unroll
    for (int mt = 0; mt < 4; ++mt)
#pragma unroll
        for (int nt = 0; nt < 2; ++nt)
            acc[mt][nt] = zero8;

    fetch(0);
    stage(0);
    __syncthreads();

    const int r  = lane & 15;           // row-within-16 / column-within-16
    const int ko = (lane >> 4) * 8;     // A K base (0 or 8); B K base = ko

    for (int kk = 0; kk < NKSTEPS; ++kk) {
        if (kk + 1 < NKSTEPS) fetch(kk + 1);   // prefetch next tile into regs

        const int cur = kk & 1;
        v16h afrag[4];
        v16h bfrag[2];
#pragma unroll
        for (int mt = 0; mt < 4; ++mt) {
            // ISA 16-bit A 16x32 layout: lane<16 holds K {0..7,16..23}, lane>=16 {8..15,24..31}
            const _Float16* p = &As[cur][(wm + mt * 16 + r) * SK + ko];
            v8h lo = *(const v8h*)(p);
            v8h hi = *(const v8h*)(p + 16);
            afrag[mt] = __builtin_shufflevector(lo, hi,
                0, 1, 2, 3, 4, 5, 6, 7, 8, 9, 10, 11, 12, 13, 14, 15);
        }
#pragma unroll
        for (int nt = 0; nt < 2; ++nt) {
            // B 32x16: lane = column N, K {0..7,16..23} (lane<16) or {8..15,24..31} (lane>=16), same as A
            const _Float16* p = &Bs[cur][(wn + nt * 16 + r) * SK + ko];
            v8h lo = *(const v8h*)(p);
            v8h hi = *(const v8h*)(p + 16);
            bfrag[nt] = __builtin_shufflevector(lo, hi,
                0, 1, 2, 3, 4, 5, 6, 7, 8, 9, 10, 11, 12, 13, 14, 15);
        }
#pragma unroll
        for (int mt = 0; mt < 4; ++mt)
#pragma unroll
            for (int nt = 0; nt < 2; ++nt)
                acc[mt][nt] = __builtin_amdgcn_wmma_f32_16x16x32_f16(
                    false, afrag[mt], false, bfrag[nt],
                    (short)0, acc[mt][nt], false, false);

        if (kk + 1 < NKSTEPS) stage((kk + 1) & 1); // fill the other buffer
        __syncthreads();
    }

    // Epilogue: C/D layout is M = vgpr + 8*(lane>=16), N = lane&15. Add bias, store f32.
#pragma unroll
    for (int nt = 0; nt < 2; ++nt) {
        const int n  = bn + wn + nt * 16 + r;
        const float bv = bias[n];
#pragma unroll
        for (int mt = 0; mt < 4; ++mt) {
            const int m0 = bm + wm + mt * 16 + (lane >> 4) * 8;
#pragma unroll
            for (int vg = 0; vg < 8; ++vg)
                out[(size_t)(m0 + vg) * OUT_F + n] = acc[mt][nt][vg] + bv;
        }
    }
}

extern "C" void kernel_launch(void* const* d_in, const int* in_sizes, int n_in,
                              void* d_out, int out_size, void* d_ws, size_t ws_size,
                              hipStream_t stream) {
    (void)n_in; (void)out_size; (void)d_ws; (void)ws_size;
    const float* x      = (const float*)d_in[0];
    const int*   qw     = (const int*)d_in[1];
    const float* scales = (const float*)d_in[2];
    const float* zeros  = (const float*)d_in[3];
    const float* bias   = (const float*)d_in[4];
    float*       out    = (float*)d_out;

    const int M = in_sizes[0] / IN_F;          // 8192 tokens
    dim3 grid(OUT_F / BN, M / BM);             // 32 x 64 = 2048 workgroups
    w4_gemm_wmma<<<grid, dim3(256), 0, stream>>>(x, qw, scales, zeros, bias, out);
}
